// GCNConvNet_9612136808894
// MI455X (gfx1250) — matrix operand order repaired and checked by reading, not compiled, not run on hardware
//
#include <hip/hip_runtime.h>
#include <hip/hip_bf16.h>
#include <cstdint>

typedef __attribute__((ext_vector_type(16))) __bf16 v16bf;
typedef __attribute__((ext_vector_type(8)))  float  v8f;

union Frag { v16bf v; uint4 q[2]; };

#define NNODES 50000

// Force the native hardware fp32 atomic (no CAS loop): device-scope,
// no-return form -> tracked by STOREcnt; S_ENDPGM's implicit wait-idle
// guarantees completion before wave exit.
static __device__ __forceinline__ void atomic_add_f32(float* p, float v) {
    asm volatile("global_atomic_add_f32 %0, %1, off scope:SCOPE_DEV"
                 :: "v"(p), "v"(v) : "memory");
}

// ---------------- degree ----------------
__global__ void k_fill(float* __restrict__ p, float v, int n) {
    int i = blockIdx.x * blockDim.x + threadIdx.x;
    if (i < n) p[i] = v;
}

__global__ void k_deg(const int* __restrict__ dst, float* __restrict__ deg, int E) {
    int e = blockIdx.x * blockDim.x + threadIdx.x;
    if (e < E) atomic_add_f32(deg + dst[e], 1.0f);
}

__global__ void k_inv(float* __restrict__ deg, int n) {
    int i = blockIdx.x * blockDim.x + threadIdx.x;
    if (i < n) deg[i] = 1.0f / deg[i];   // deg >= 1 always (self loop)
}

__global__ void k_copy(const float* __restrict__ s, float* __restrict__ d, int n) {
    int i = blockIdx.x * blockDim.x + threadIdx.x;
    if (i < n) d[i] = s[i];
}

// ---------------- weight packing into WMMA-B fragment order (bf16) ----------------
// B (KPADx16 per tile) lane layout (ISA 7.12.2 / sparse-B analog):
//   lane L: column n = L&15 ; K values = kt*32 + (L>>4)*8 + (e&7) + (e>>3)*16 , e = 0..15.
__global__ void k_pack_w(const float* __restrict__ W, unsigned short* __restrict__ P,
                         int K, int Nvalid, int NKT, int total) {
    int t = blockIdx.x * blockDim.x + threadIdx.x;
    if (t >= total) return;
    int e    = t & 15;
    int lane = (t >> 4) & 31;
    int tp   = t >> 9;            // nt*NKT + kt
    int kt   = tp % NKT;
    int nt   = tp / NKT;
    int n    = nt * 16 + (lane & 15);
    int k    = kt * 32 + ((lane >> 4) << 3) + (e & 7) + ((e >> 3) << 4);
    float v = 0.0f;
    if (k < K && n < Nvalid) v = W[k * Nvalid + n];
    ((__bf16*)P)[t] = (__bf16)v;
}

// ---------------- edge aggregation (scatter-add) ----------------
__global__ void k_scatter3(const int* __restrict__ src, const int* __restrict__ dst,
                           const float* __restrict__ X, float* __restrict__ AGG, int E) {
    int e = blockIdx.x * blockDim.x + threadIdx.x;
    if (e >= E) return;
    int s = src[e], d = dst[e];
    atomic_add_f32(AGG + (size_t)d * 3 + 0, X[(size_t)s * 3 + 0]);
    atomic_add_f32(AGG + (size_t)d * 3 + 1, X[(size_t)s * 3 + 1]);
    atomic_add_f32(AGG + (size_t)d * 3 + 2, X[(size_t)s * 3 + 2]);
}

template<int SH> // F = 4 << SH ; chunks of float4 per edge = 1<<SH
__global__ void k_scatter_vec(const int* __restrict__ src, const int* __restrict__ dst,
                              const float* __restrict__ X, float* __restrict__ AGG, int E) {
    long long t = (long long)blockIdx.x * blockDim.x + threadIdx.x;
    int e = (int)(t >> SH);
    if (e >= E) return;
    int c = (int)t & ((1 << SH) - 1);
    constexpr int F = 4 << SH;
    int s = src[e], d = dst[e];
    const float* xp = X + (size_t)s * F + c * 4;
    __builtin_prefetch(xp, 0, 0);                 // global_prefetch_b8 (cacheline hint)
    const float4 v = *(const float4*)xp;
    float* p = AGG + (size_t)d * F + c * 4;
    atomic_add_f32(p + 0, v.x);
    atomic_add_f32(p + 1, v.y);
    atomic_add_f32(p + 2, v.z);
    atomic_add_f32(p + 3, v.w);
}

// ---------------- fused dual GEMM:  H = ReLU( (dinv*AGG)@Wout + b + X@Wroot ) ----------------
// block = 128 threads (4 waves); blockIdx.x = M-tile (16 rows); wave w -> ntile blockIdx.y*4+w
template<int KPAD>
__global__ void __launch_bounds__(128)
k_gemm_dual(const float* __restrict__ AGG, const float* __restrict__ X,
            const float* __restrict__ dinv,
            const unsigned short* __restrict__ Bout, const unsigned short* __restrict__ Broot,
            const float* __restrict__ bias, float* __restrict__ H,
            int K, int N) {
    constexpr int NKT = KPAD / 32;
    constexpr int LDP = KPAD + 8;                 // +16B row pad: kill LDS bank conflicts
    __shared__ __bf16 Abuf[16 * LDP];
    __shared__ __bf16 Xbuf[16 * LDP];

    const int tid = threadIdx.x;
    const int r0  = blockIdx.x * 16;

    for (int i = tid; i < 16 * KPAD; i += 128) {
        int m = i / KPAD;
        int k = i % KPAD;
        int row = r0 + m;
        float a = 0.0f, xv = 0.0f;
        if (k < K) {
            a  = AGG[(size_t)row * K + k] * dinv[row];
            xv = X[(size_t)row * K + k];
        }
        Abuf[m * LDP + k] = (__bf16)a;
        Xbuf[m * LDP + k] = (__bf16)xv;
    }
    __syncthreads();

    const int lane = tid & 31;
    const int wv   = tid >> 5;
    const int nt   = blockIdx.y * 4 + wv;
    const int nl   = lane & 15;
    const int n    = nt * 16 + nl;
    const int kb   = (lane >> 4) << 3;            // A-frag K base: 0 or 8

    const float bb = bias[n];
    v8f c = { bb, bb, bb, bb, bb, bb, bb, bb };

    const uint4* bo = (const uint4*)(Bout  + (size_t)nt * NKT * 512);
    const uint4* br = (const uint4*)(Broot + (size_t)nt * NKT * 512);

    #pragma unroll
    for (int kt = 0; kt < NKT; ++kt) {
        Frag fa, fx, fwo, fwr;
        const __bf16* ap = Abuf + nl * LDP + kt * 32 + kb;
        fa.q[0] = *(const uint4*)(ap);
        fa.q[1] = *(const uint4*)(ap + 16);
        const __bf16* xp = Xbuf + nl * LDP + kt * 32 + kb;
        fx.q[0] = *(const uint4*)(xp);
        fx.q[1] = *(const uint4*)(xp + 16);
        fwo.q[0] = bo[(kt * 32 + lane) * 2 + 0];
        fwo.q[1] = bo[(kt * 32 + lane) * 2 + 1];
        fwr.q[0] = br[(kt * 32 + lane) * 2 + 0];
        fwr.q[1] = br[(kt * 32 + lane) * 2 + 1];
        c = __builtin_amdgcn_wmma_f32_16x16x32_bf16(false, fa.v, false, fwo.v,
                                                    (short)0, c, false, false);
        c = __builtin_amdgcn_wmma_f32_16x16x32_bf16(false, fx.v, false, fwr.v,
                                                    (short)0, c, false, false);
    }

    const int mb = (lane >> 4) << 3;              // C rows: lanes<16 -> 0..7, lanes>=16 -> 8..15
    #pragma unroll
    for (int v = 0; v < 8; ++v) {
        float o = c[v];
        o = o < 0.0f ? 0.0f : o;                  // ReLU
        H[(size_t)(r0 + mb + v) * N + n] = o;
    }
}

// ---------------- head:  OUT = sigmoid( h3 @ W_lin + b_lin ), K=256, N=16 (6 valid) ----------------
__global__ void __launch_bounds__(32)
k_gemm_head(const float* __restrict__ A, const unsigned short* __restrict__ Bw,
            const float* __restrict__ bias, float* __restrict__ OUT) {
    const int lane = threadIdx.x;
    const int r0   = blockIdx.x * 16;
    const int nl   = lane & 15;
    const int kb   = (lane >> 4) << 3;

    const float bb = (nl < 6) ? bias[nl] : 0.0f;
    v8f c = { bb, bb, bb, bb, bb, bb, bb, bb };

    const float* arow = A + (size_t)(r0 + nl) * 256;
    const uint4* bq = (const uint4*)Bw;

    #pragma unroll
    for (int kt = 0; kt < 8; ++kt) {
        int k0 = kt * 32 + kb;
        float4 f0 = *(const float4*)(arow + k0);
        float4 f1 = *(const float4*)(arow + k0 + 4);
        float4 f2 = *(const float4*)(arow + k0 + 16);
        float4 f3 = *(const float4*)(arow + k0 + 20);
        Frag fa, fw;
        fa.v[0] = (__bf16)f0.x; fa.v[1] = (__bf16)f0.y; fa.v[2] = (__bf16)f0.z; fa.v[3] = (__bf16)f0.w;
        fa.v[4] = (__bf16)f1.x; fa.v[5] = (__bf16)f1.y; fa.v[6] = (__bf16)f1.z; fa.v[7] = (__bf16)f1.w;
        fa.v[8]  = (__bf16)f2.x; fa.v[9]  = (__bf16)f2.y; fa.v[10] = (__bf16)f2.z; fa.v[11] = (__bf16)f2.w;
        fa.v[12] = (__bf16)f3.x; fa.v[13] = (__bf16)f3.y; fa.v[14] = (__bf16)f3.z; fa.v[15] = (__bf16)f3.w;
        fw.q[0] = bq[(kt * 32 + lane) * 2 + 0];
        fw.q[1] = bq[(kt * 32 + lane) * 2 + 1];
        c = __builtin_amdgcn_wmma_f32_16x16x32_bf16(false, fa.v, false, fw.v,
                                                    (short)0, c, false, false);
    }

    if (nl < 6) {
        const int mb = (lane >> 4) << 3;
        #pragma unroll
        for (int v = 0; v < 8; ++v) {
            float s = 1.0f / (1.0f + __expf(-c[v]));
            OUT[(size_t)(r0 + mb + v) * 6 + nl] = s;
        }
    }
}

// ---------------- host ----------------
extern "C" void kernel_launch(void* const* d_in, const int* in_sizes, int n_in,
                              void* d_out, int out_size, void* d_ws, size_t ws_size,
                              hipStream_t stream) {
    const float* x   = (const float*)d_in[0];
    const int*   ei  = (const int*)d_in[1];
    const int    E   = in_sizes[1] / 2;
    const int*   src = ei;
    const int*   dst = ei + E;
    const float* W1o = (const float*)d_in[2];
    const float* b1  = (const float*)d_in[3];
    const float* W1r = (const float*)d_in[4];
    const float* W2o = (const float*)d_in[5];
    const float* b2  = (const float*)d_in[6];
    const float* W2r = (const float*)d_in[7];
    const float* W3o = (const float*)d_in[8];
    const float* b3  = (const float*)d_in[9];
    const float* W3r = (const float*)d_in[10];
    const float* Wl  = (const float*)d_in[11];
    const float* bl  = (const float*)d_in[12];
    float* out = (float*)d_out;

    const int NN = NNODES;

    // workspace carve-out (256B aligned)
    char* base = (char*)d_ws;
    size_t off = 0;
    auto carve = [&](size_t bytes) -> void* {
        void* p = base + off;
        off = (off + bytes + 255) & ~(size_t)255;
        return p;
    };
    float* deg = (float*)carve((size_t)NN * 4);
    float* agg = (float*)carve((size_t)NN * 128 * 4);
    float* h1  = (float*)carve((size_t)NN * 64 * 4);
    float* h2  = (float*)carve((size_t)NN * 128 * 4);
    float* h3  = (float*)carve((size_t)NN * 256 * 4);
    unsigned short* p1o = (unsigned short*)carve(2048 * 2);
    unsigned short* p1r = (unsigned short*)carve(2048 * 2);
    unsigned short* p2o = (unsigned short*)carve(8192 * 2);
    unsigned short* p2r = (unsigned short*)carve(8192 * 2);
    unsigned short* p3o = (unsigned short*)carve(32768 * 2);
    unsigned short* p3r = (unsigned short*)carve(32768 * 2);
    unsigned short* plin = (unsigned short*)carve(4096 * 2);
    (void)ws_size; (void)n_in; (void)out_size;

    auto blocks = [](long long work, int bs) { return (unsigned)((work + bs - 1) / bs); };

    // ---- degrees (self-loop folded into init = 1.0) ----
    k_fill<<<blocks(NN, 256), 256, 0, stream>>>(deg, 1.0f, NN);
    k_deg<<<blocks(E, 256), 256, 0, stream>>>(dst, deg, E);
    k_inv<<<blocks(NN, 256), 256, 0, stream>>>(deg, NN);

    // ---- pack weights into WMMA-B fragment layout (bf16, zero padded) ----
    auto pack = [&](const float* W, unsigned short* P, int K, int Nvalid, int Npad, int KPAD) {
        int NKT = KPAD / 32;
        int total = (Npad / 16) * NKT * 512;
        k_pack_w<<<blocks(total, 256), 256, 0, stream>>>(W, P, K, Nvalid, NKT, total);
    };
    pack(W1o, p1o, 3, 64, 64, 32);
    pack(W1r, p1r, 3, 64, 64, 32);
    pack(W2o, p2o, 64, 128, 128, 64);
    pack(W2r, p2r, 64, 128, 128, 64);
    pack(W3o, p3o, 128, 256, 256, 128);
    pack(W3r, p3r, 128, 256, 256, 128);
    pack(Wl, plin, 256, 6, 16, 256);

    const unsigned MT = NN / 16; // 3125 M-tiles

    // ---- layer 1: F 3 -> 64 ----
    k_copy<<<blocks((long long)NN * 3, 256), 256, 0, stream>>>(x, agg, NN * 3);
    k_scatter3<<<blocks(E, 256), 256, 0, stream>>>(src, dst, x, agg, E);
    k_gemm_dual<32><<<dim3(MT, 1), 128, 0, stream>>>(agg, x, deg, p1o, p1r, b1, h1, 3, 64);

    // ---- layer 2: F 64 -> 128 ----
    k_copy<<<blocks((long long)NN * 64, 256), 256, 0, stream>>>(h1, agg, NN * 64);
    k_scatter_vec<4><<<blocks((long long)E << 4, 256), 256, 0, stream>>>(src, dst, h1, agg, E);
    k_gemm_dual<64><<<dim3(MT, 2), 128, 0, stream>>>(agg, h1, deg, p2o, p2r, b2, h2, 64, 128);

    // ---- layer 3: F 128 -> 256 ----
    k_copy<<<blocks((long long)NN * 128, 256), 256, 0, stream>>>(h2, agg, NN * 128);
    k_scatter_vec<5><<<blocks((long long)E << 5, 256), 256, 0, stream>>>(src, dst, h2, agg, E);
    k_gemm_dual<128><<<dim3(MT, 4), 128, 0, stream>>>(agg, h2, deg, p3o, p3r, b3, h3, 128, 256);

    // ---- head: 256 -> 6, sigmoid ----
    k_gemm_head<<<MT, 32, 0, stream>>>(h3, plin, bl, out);
}
